// GraphAttentionReadout_4647154614310
// MI455X (gfx1250) — matrix operand order repaired and checked by reading, not compiled, not run on hardware
//
#include <hip/hip_runtime.h>
#include <hip/hip_bf16.h>
#include <stdint.h>

// CDNA5 / gfx1250 WMMA + packed vector types
typedef __attribute__((ext_vector_type(16))) __bf16 v16bf;
typedef __attribute__((ext_vector_type(8)))  float  v8f;
typedef __attribute__((ext_vector_type(16))) float  v16f;
typedef __attribute__((ext_vector_type(4)))  float  v4f;
typedef __attribute__((ext_vector_type(2)))  float  v2f;   // -> v_pk_fma_f32

#define NB 128    // batch
#define NN 2048   // nodes
#define QD 256    // query dim
#define ND 128    // node dim
#define OD 256    // out dim
#define NTHREADS 512
#define NWAVE (NTHREADS / 32)

union V16F { v16f v; float4 q[4]; float f[16]; };

__device__ __forceinline__ float wave_max(float v) {
#pragma unroll
    for (int m = 16; m >= 1; m >>= 1) v = fmaxf(v, __shfl_xor(v, m, 32));
    return v;
}
__device__ __forceinline__ float wave_sum(float v) {
#pragma unroll
    for (int m = 16; m >= 1; m >>= 1) v += __shfl_xor(v, m, 32);
    return v;
}
// Broadcast value from (compile-time-uniform) lane l -> SGPR, feeds pk-FMA as scalar.
__device__ __forceinline__ float bcast(float v, int l) {
    return __uint_as_float(__builtin_amdgcn_readlane(__float_as_uint(v), l));
}

__global__ __launch_bounds__(NTHREADS)
void GraphAttentionReadout_4647154614310_kernel(
    const float* __restrict__ fp_vec,       // (B, QD)
    const float* __restrict__ node_mat,     // (B, NN, ND)
    const unsigned char* __restrict__ mask, // (B, NN) bool
    const float* __restrict__ Wq1,          // (ND, QD)
    const float* __restrict__ Wq2,          // (ND, QD)
    const float* __restrict__ Wk,           // (ND, ND)
    const float* __restrict__ Wv,           // (OD, ND)
    const float* __restrict__ lambda_logit,
    const float* __restrict__ gamma,
    const float* __restrict__ beta,
    float* __restrict__ out)                // (B, OD)
{
    __shared__ __align__(16) float fp_s[QD];
    __shared__ float q_s[2][ND];
    __shared__ __align__(16) float qk_s[2][ND];
    __shared__ float lam_s;
    __shared__ float accbuf[NWAVE][2][ND];
    __shared__ float4 mstat[NWAVE];
    __shared__ __align__(16) float tdiff_s[ND];
    __shared__ float zred[8][2];

    const int b    = blockIdx.x;
    const int t    = threadIdx.x;
    const int lane = t & 31;
    const int wave = t >> 5;

    // ---------------- Phase 0: q = fp @ Wq.T ; qk = q @ Wk ; lambda --------
    if (t < QD) fp_s[t] = fp_vec[(size_t)b * QD + t];
    if (t == 0) {
        float x = lambda_logit[0];
        lam_s = 0.8f + 0.2f / (1.0f + __expf(-x));
    }
    __syncthreads();

    if (t < 256) {
        const int which = t >> 7;      // 0 -> q1, 1 -> q2
        const int e = t & 127;
        const float* Wq = which ? Wq2 : Wq1;
        const float4* wr = (const float4*)(Wq + (size_t)e * QD);
        const float4* fr = (const float4*)fp_s;
        float s = 0.f;
#pragma unroll 8
        for (int d4 = 0; d4 < QD / 4; ++d4) {
            float4 w = wr[d4]; float4 f = fr[d4];
            s += w.x * f.x + w.y * f.y + w.z * f.z + w.w * f.w;
        }
        q_s[which][e] = s;
    }
    __syncthreads();

    if (t < 256) {
        const int which = t >> 7;
        const int d = t & 127;
        const float* qv = q_s[which];
        float s = 0.f;
#pragma unroll 8
        for (int e = 0; e < ND; ++e) s += qv[e] * Wk[(size_t)e * ND + d];
        qk_s[which][d] = s;
    }
    __syncthreads();

    // ---------------- Build WMMA A operand (rows 0/1 = qk1/qk2, bf16) ------
    // A packed here in contiguous k order (same as B): lanes 0-15 hold row m, K = c*32 + {0..15};
    // lanes 16-31 hold row m, K = c*32 + {16..31}.
    const int r  = lane & 15;
    const int hi = lane >> 4;
    v16bf av[4];
#pragma unroll
    for (int c = 0; c < 4; ++c) {
        V16F xf;
#pragma unroll
        for (int i = 0; i < 16; ++i) {
            int k = c * 32 + hi * 16 + i;
            xf.f[i] = (r == 0) ? qk_s[0][k] : ((r == 1) ? qk_s[1][k] : 0.f);
        }
        av[c] = __builtin_convertvector(xf.v, v16bf);   // v_cvt_pk_bf16_f32 x8
    }

    // ---------------- Phase 1: single pass, online softmax x2 --------------
    const float inv_scale = 0.08838834764831845f;   // 1/sqrt(128)
    float m1 = -1e30f, l1 = 0.f, m2 = -1e30f, l2 = 0.f;
    v2f acc1a = {0.f, 0.f}, acc1b = {0.f, 0.f};
    v2f acc2a = {0.f, 0.f}, acc2b = {0.f, 0.f};
    const size_t rowbase = (size_t)b * NN;

    for (int tile = wave; tile < NN / 16; tile += NWAVE) {
        const int n0 = tile * 16;
        // B operand: 16x32 bf16 slices of node^T; lane holds node row (lane&15),
        // K slice c*32 + (lane>>4)*16 .. +16 (sequential packing).
        const float* rowp = node_mat + (rowbase + n0 + r) * (size_t)ND + hi * 16;
        __builtin_prefetch(rowp + (size_t)NWAVE * 16 * ND, 0, 1);  // next tile for this wave

        // Stage all loads, then all converts (distinct regs), then two
        // independent WMMA accumulation chains -> minimal hazard NOPs.
        V16F xf0, xf1, xf2, xf3;
        {
            const float4* p0 = (const float4*)(rowp);
            const float4* p1 = (const float4*)(rowp + 32);
            const float4* p2 = (const float4*)(rowp + 64);
            const float4* p3 = (const float4*)(rowp + 96);
            xf0.q[0] = p0[0]; xf0.q[1] = p0[1]; xf0.q[2] = p0[2]; xf0.q[3] = p0[3];
            xf1.q[0] = p1[0]; xf1.q[1] = p1[1]; xf1.q[2] = p1[2]; xf1.q[3] = p1[3];
            xf2.q[0] = p2[0]; xf2.q[1] = p2[1]; xf2.q[2] = p2[2]; xf2.q[3] = p2[3];
            xf3.q[0] = p3[0]; xf3.q[1] = p3[1]; xf3.q[2] = p3[2]; xf3.q[3] = p3[3];
        }
        v16bf bop0 = __builtin_convertvector(xf0.v, v16bf);
        v16bf bop1 = __builtin_convertvector(xf1.v, v16bf);
        v16bf bop2 = __builtin_convertvector(xf2.v, v16bf);
        v16bf bop3 = __builtin_convertvector(xf3.v, v16bf);

        v8f D0 = {};
        v8f D1 = {};
        D0 = __builtin_amdgcn_wmma_f32_16x16x32_bf16(false, av[0], false, bop0, (short)0, D0, false, false);
        D1 = __builtin_amdgcn_wmma_f32_16x16x32_bf16(false, av[1], false, bop1, (short)0, D1, false, false);
        D0 = __builtin_amdgcn_wmma_f32_16x16x32_bf16(false, av[2], false, bop2, (short)0, D0, false, false);
        D1 = __builtin_amdgcn_wmma_f32_16x16x32_bf16(false, av[3], false, bop3, (short)0, D1, false, false);

        // lane n (<16): row0 = s1[node n], row1 = s2[node n]
        float s1 = (D0[0] + D1[0]) * inv_scale;
        float s2 = (D0[1] + D1[1]) * inv_scale;
        unsigned char mk = mask[rowbase + n0 + r];
        float pad = mk ? 0.f : -1e9f;
        s1 += pad; s2 += pad;
        if (lane >= 16) { s1 = -1e30f; s2 = -1e30f; }

        float tm1 = wave_max(s1), tm2 = wave_max(s2);
        float nm1 = fmaxf(m1, tm1), nm2 = fmaxf(m2, tm2);
        float sc1 = __expf(m1 - nm1), sc2 = __expf(m2 - nm2);
        float p1  = __expf(s1 - nm1), p2  = __expf(s2 - nm2); // lanes>=16 -> 0
        l1 = l1 * sc1 + wave_sum(p1);
        l2 = l2 * sc2 + wave_sum(p2);
        m1 = nm1; m2 = nm2;
        acc1a *= sc1; acc1b *= sc1;        // v_pk_mul_f32
        acc2a *= sc2; acc2b *= sc2;

        // Weighted accumulation: lane covers d = 4*lane .. 4*lane+3
        const float* colp = node_mat + (rowbase + n0) * (size_t)ND + lane * 4;
#pragma unroll
        for (int j = 0; j < 16; ++j) {
            float w1 = bcast(p1, j);       // v_readlane -> SGPR operand
            float w2 = bcast(p2, j);
            v4f nv = *(const v4f*)(colp + (size_t)j * ND);
            v2f lo = {nv[0], nv[1]};
            v2f hs = {nv[2], nv[3]};
            acc1a += w1 * lo;              // v_pk_fma_f32
            acc1b += w1 * hs;
            acc2a += w2 * lo;
            acc2b += w2 * hs;
        }
    }

    // ---------------- Phase 2: merge wave states (flash combine) -----------
    accbuf[wave][0][lane * 4 + 0] = acc1a[0];
    accbuf[wave][0][lane * 4 + 1] = acc1a[1];
    accbuf[wave][0][lane * 4 + 2] = acc1b[0];
    accbuf[wave][0][lane * 4 + 3] = acc1b[1];
    accbuf[wave][1][lane * 4 + 0] = acc2a[0];
    accbuf[wave][1][lane * 4 + 1] = acc2a[1];
    accbuf[wave][1][lane * 4 + 2] = acc2b[0];
    accbuf[wave][1][lane * 4 + 3] = acc2b[1];
    if (lane == 0) mstat[wave] = make_float4(m1, l1, m2, l2);
    __syncthreads();

    if (t < ND) {
        float gm1 = -1e30f, gm2 = -1e30f;
#pragma unroll
        for (int w = 0; w < NWAVE; ++w) {
            float4 s = mstat[w];
            gm1 = fmaxf(gm1, s.x);
            gm2 = fmaxf(gm2, s.z);
        }
        float L1 = 0.f, L2 = 0.f, t1 = 0.f, t2 = 0.f;
#pragma unroll
        for (int w = 0; w < NWAVE; ++w) {
            float4 s = mstat[w];
            float e1 = __expf(s.x - gm1), e2 = __expf(s.z - gm2);
            L1 += s.y * e1; L2 += s.w * e2;
            t1 += accbuf[w][0][t] * e1;
            t2 += accbuf[w][1][t] * e2;
        }
        tdiff_s[t] = t1 / L1 - lam_s * (t2 / L2);
    }
    __syncthreads();

    // ---------------- Phase 3: z = Wv @ tdiff ; LayerNorm ------------------
    float z = 0.f;
    if (t < OD) {
        const float4* wr = (const float4*)(Wv + (size_t)t * ND);
        const float4* td = (const float4*)tdiff_s;
#pragma unroll 8
        for (int d4 = 0; d4 < ND / 4; ++d4) {
            float4 w = wr[d4]; float4 x = td[d4];
            z += w.x * x.x + w.y * x.y + w.z * x.z + w.w * x.w;
        }
    }
    float zs = wave_sum(z);
    float zq = wave_sum(z * z);
    if (lane == 0 && wave < 8) { zred[wave][0] = zs; zred[wave][1] = zq; }
    __syncthreads();

    if (t < OD) {
        float S = 0.f, Q = 0.f;
#pragma unroll
        for (int w = 0; w < 8; ++w) { S += zred[w][0]; Q += zred[w][1]; }
        float mu  = S / OD;
        float var = Q / OD - mu * mu;
        out[(size_t)b * OD + t] = (z - mu) * rsqrtf(var + 1e-5f) * gamma[t] + beta[t];
    }
}

extern "C" void kernel_launch(void* const* d_in, const int* in_sizes, int n_in,
                              void* d_out, int out_size, void* d_ws, size_t ws_size,
                              hipStream_t stream) {
    const float* fp_vec       = (const float*)d_in[0];
    const float* node_mat     = (const float*)d_in[1];
    const unsigned char* mask = (const unsigned char*)d_in[2];
    const float* Wq1          = (const float*)d_in[3];
    const float* Wq2          = (const float*)d_in[4];
    const float* Wk           = (const float*)d_in[5];
    const float* Wv           = (const float*)d_in[6];
    const float* lambda_logit = (const float*)d_in[7];
    const float* gamma        = (const float*)d_in[8];
    const float* beta         = (const float*)d_in[9];
    float* out                = (float*)d_out;

    GraphAttentionReadout_4647154614310_kernel<<<NB, NTHREADS, 0, stream>>>(
        fp_vec, node_mat, mask, Wq1, Wq2, Wk, Wv, lambda_logit, gamma, beta, out);
}
